// MVTS_GCN_RNN_84937273246207
// MI455X (gfx1250) — matrix operand order repaired and checked by reading, not compiled, not run on hardware
//
#include <hip/hip_runtime.h>
#include <hip/hip_bf16.h>
#include <cstddef>

// ---------------------------------------------------------------------------
// MVTS_GCN_RNN for MI455X (gfx1250, wave32).
// - all matmuls: v_wmma_f32_16x16x32_bf16
// - GEMM tile staging: global_load_async_to_lds_b128 (ASYNCcnt) + double buffer
// - operands pre-converted to bf16 once; GEMM K-loop moves raw bytes only
// ---------------------------------------------------------------------------

#define TT  8
#define NN  2048
#define DD  1024
#define SED 512

typedef __attribute__((ext_vector_type(16))) __bf16          v16bf;
typedef __attribute__((ext_vector_type(8)))  float           v8f;
typedef __attribute__((ext_vector_type(8)))  unsigned short  v8us;

union F16 { v16bf v; v8us h[2]; unsigned short u[16]; };

__device__ inline unsigned short f2bf(float f) {             // RNE truncate
  unsigned int x = __float_as_uint(f);
  unsigned int r = x + 0x7FFFu + ((x >> 16) & 1u);
  return (unsigned short)(r >> 16);
}
__device__ inline float bf2f(unsigned short u) {
  return __uint_as_float((unsigned int)u << 16);
}
__device__ inline float sigm(float x) { return 1.0f / (1.0f + expf(-x)); }

__device__ inline v8f wmma_bf16(const F16& a, const F16& b, v8f c) {
  return __builtin_amdgcn_wmma_f32_16x16x32_bf16(false, a.v, false, b.v,
                                                 (short)0, c, false, false);
}

// Async copy 16B/lane from global (saddr + voff) into LDS byte offset.
__device__ inline void async_b128(unsigned lds_off, unsigned gvoff, const void* sbase) {
  asm volatile("global_load_async_to_lds_b128 %0, %1, %2"
               :: "v"(lds_off), "v"(gvoff), "s"(sbase) : "memory");
}
#define WAIT_ASYNC_4() asm volatile("s_wait_asynccnt 4" ::: "memory")
#define WAIT_ASYNC_0() asm volatile("s_wait_asynccnt 0" ::: "memory")

// ---------------------------------------------------------------------------
// GEMM: C[M,N] = A[M,K](bf16) * BT[N,K](bf16)^T  (+bias)(+relu)
// outMode: 0 = f32 row-major C (ldc), 1 = bf16 row-major C, 2 = bf16 C^T (ldc
// is leading dim of the transposed layout).  M%128==0, N%128==0, K%64==0.
// 256 threads = 8 waves; tile 128x128x32; double-buffered async LDS staging.
// ---------------------------------------------------------------------------
__global__ __launch_bounds__(256) void k_gemm_async(
    const unsigned short* __restrict__ A, const unsigned short* __restrict__ BT,
    const float* __restrict__ bias, void* __restrict__ C,
    int K, int lda, int ldbt, int ldc, int outMode, int doRelu)
{
  constexpr int LDT  = 40;               // bf16 elems per LDS tile row (80B)
  constexpr int TILE = 128 * LDT;        // 5120 elems = 10240 B
  __shared__ alignas(16) unsigned short smem[4 * TILE];  // A0 B0 A1 B1

  const int tid  = threadIdx.x;
  const int row0 = blockIdx.y * 128, col0 = blockIdx.x * 128;
  const int w    = tid >> 5, lane = tid & 31;
  const int wr   = w >> 2,  wc   = w & 3;          // 2 x 4 wave grid
  const int mlo  = lane & 15, half = lane >> 4;
  const unsigned ldsBase = (unsigned)(size_t)&smem[0];

  v8f acc[4][2];
  const v8f zf = {0.f, 0.f, 0.f, 0.f, 0.f, 0.f, 0.f, 0.f};
  #pragma unroll
  for (int i = 0; i < 4; ++i) { acc[i][0] = zf; acc[i][1] = zf; }

  auto issue = [&](int kt, int buf) {
    unsigned aBase = ldsBase + (unsigned)(buf * 2 * TILE) * 2u;
    unsigned bBase = aBase + (unsigned)TILE * 2u;
    #pragma unroll
    for (int i = 0; i < 2; ++i) {                  // A: 512 x 16B chunks
      int q = tid + i * 256;
      int r = q >> 2, c = q & 3;
      async_b128(aBase + (unsigned)(r * LDT + c * 8) * 2u,
                 (unsigned)(((row0 + r) * lda + kt + c * 8) * 2), A);
    }
    #pragma unroll
    for (int i = 0; i < 2; ++i) {                  // BT: 512 x 16B chunks
      int q = tid + i * 256;
      int n = q >> 2, c = q & 3;
      async_b128(bBase + (unsigned)(n * LDT + c * 8) * 2u,
                 (unsigned)(((col0 + n) * ldbt + kt + c * 8) * 2), BT);
    }
  };

  const int nIter = K >> 5;
  issue(0, 0);
  for (int it = 0; it < nIter; ++it) {
    const int cur = it & 1;
    if (it + 1 < nIter) { issue((it + 1) << 5, cur ^ 1); WAIT_ASYNC_4(); }
    else                { WAIT_ASYNC_0(); }
    __syncthreads();                               // tile `it` visible to all

    const unsigned short* Ab = smem + cur * 2 * TILE;
    const unsigned short* Bb = Ab + TILE;

    F16 fb[2];                                     // B frag: k = half*16 + e
    #pragma unroll
    for (int ni = 0; ni < 2; ++ni) {
      const unsigned short* p = Bb + (wc * 32 + ni * 16 + mlo) * LDT + half * 16;
      fb[ni].h[0] = *(const v8us*)p;
      fb[ni].h[1] = *(const v8us*)(p + 8);
    }
    #pragma unroll
    for (int mi = 0; mi < 4; ++mi) {               // A frag: k = half*16 + e
      F16 fa;
      const unsigned short* p = Ab + (wr * 64 + mi * 16 + mlo) * LDT + half * 16;
      fa.h[0] = *(const v8us*)p;
      fa.h[1] = *(const v8us*)(p + 8);
      acc[mi][0] = wmma_bf16(fa, fb[0], acc[mi][0]);
      acc[mi][1] = wmma_bf16(fa, fb[1], acc[mi][1]);
    }
    __syncthreads();                               // done reading buf `cur`
  }

  // epilogue: lane holds (row = r + 8*half, col = mlo) of each 16x16 tile
  #pragma unroll
  for (int mi = 0; mi < 4; ++mi)
    #pragma unroll
    for (int ni = 0; ni < 2; ++ni) {
      int gc = col0 + wc * 32 + ni * 16 + mlo;
      float bv = bias ? bias[gc] : 0.f;
      int gr0 = row0 + wr * 64 + mi * 16 + half * 8;
      if (outMode == 2) {                          // bf16 transposed: 16B store
        v8us pk;
        #pragma unroll
        for (int r = 0; r < 8; ++r) {
          float v = acc[mi][ni][r] + bv;
          if (doRelu) v = fmaxf(v, 0.f);
          pk[r] = f2bf(v);
        }
        *(v8us*)((unsigned short*)C + (size_t)gc * ldc + gr0) = pk;
      } else if (outMode == 1) {                   // bf16 row-major
        #pragma unroll
        for (int r = 0; r < 8; ++r) {
          float v = acc[mi][ni][r] + bv;
          if (doRelu) v = fmaxf(v, 0.f);
          ((unsigned short*)C)[(size_t)(gr0 + r) * ldc + gc] = f2bf(v);
        }
      } else {                                     // f32 row-major
        #pragma unroll
        for (int r = 0; r < 8; ++r) {
          float v = acc[mi][ni][r] + bv;
          if (doRelu) v = fmaxf(v, 0.f);
          ((float*)C)[(size_t)(gr0 + r) * ldc + gc] = v;
        }
      }
    }
}

// ---------------------------------------------------------------------------
// Converters / producers (all emit bf16 operands for the GEMMs)
// ---------------------------------------------------------------------------
__global__ void k_cvt(const float* __restrict__ s, unsigned short* __restrict__ d, int n)
{
  int i = blockIdx.x * blockDim.x + threadIdx.x;
  if (i < n) d[i] = f2bf(s[i]);
}

// d[C,R] = bf16(s[R,C]^T)
__global__ void k_cvtT(const float* __restrict__ s, unsigned short* __restrict__ d,
                       int R, int C)
{
  __shared__ float t[32][33];
  int r0 = blockIdx.y * 32, c0 = blockIdx.x * 32;
  int tx = threadIdx.x, ty = threadIdx.y;
  t[ty][tx] = s[(size_t)(r0 + ty) * C + c0 + tx];
  __syncthreads();
  d[(size_t)(c0 + ty) * R + r0 + tx] = f2bf(t[tx][ty]);
}

// X1bf[(d*8+t), n] = bf16(att[t, n, d])
__global__ void k_transpose_att(const float* __restrict__ att,
                                unsigned short* __restrict__ X1)
{
  __shared__ float s[32][33];
  int t = blockIdx.z;
  int n0 = blockIdx.x * 32, d0 = blockIdx.y * 32;
  int tx = threadIdx.x, ty = threadIdx.y;
  s[ty][tx] = att[(size_t)t * NN * DD + (size_t)(n0 + ty) * DD + d0 + tx];
  __syncthreads();
  X1[(size_t)((d0 + ty) * 8 + t) * NN + n0 + tx] = f2bf(s[tx][ty]);
}

// ---------------------------------------------------------------------------
// GCN normalization
// ---------------------------------------------------------------------------
__global__ void k_deg(const float* __restrict__ adj, float* __restrict__ dinv)
{
  int j = blockIdx.x * blockDim.x + threadIdx.x;       // 2048 threads
  float s = 0.f;
  for (int i = 0; i < NN; ++i) s += adj[(size_t)i * NN + j];
  float dg = adj[(size_t)j * NN + j];
  s += (dg > 0.f) ? 0.f : 1.f;                          // self-loop in colsum
  dinv[j] = (s > 0.f) ? rsqrtf(s) : 0.f;
}

// nT[i,k] = bf16(dinv[i]*dinv[k]*Ahat[k,i])
__global__ void k_normT(const float* __restrict__ adj, const float* __restrict__ dinv,
                        unsigned short* __restrict__ nT)
{
  __shared__ float s[32][33];
  int i0 = blockIdx.y * 32, k0 = blockIdx.x * 32;
  int tx = threadIdx.x, ty = threadIdx.y;
  int kk = k0 + ty, ii = i0 + tx;
  float v = adj[(size_t)kk * NN + ii];
  if (kk == ii && !(v > 0.f)) v += 1.f;                 // add_remaining_self_loops
  s[ty][tx] = v;
  __syncthreads();
  int oi = i0 + ty, ok = k0 + tx;
  nT[(size_t)oi * NN + ok] = f2bf(dinv[oi] * dinv[ok] * s[tx][ty]);
}

// graph pool: dst[j] = mean_i H2T[j, i]   (H2T is bf16 [512, 2048])
__global__ void k_pool(const unsigned short* __restrict__ H2T, float* __restrict__ dst)
{
  int j = blockIdx.x * blockDim.x + threadIdx.x;        // 512 threads
  float s = 0.f;
  const unsigned short* row = H2T + (size_t)j * NN;
  for (int i = 0; i < NN; ++i) s += bf2f(row[i]);
  dst[j] = s * (1.f / (float)NN);
}

// ---------------------------------------------------------------------------
// LSTM-1 setup
// ---------------------------------------------------------------------------
__global__ void k_biassum(const float* a, const float* b, float* o, int n)
{
  int i = blockIdx.x * blockDim.x + threadIdx.x;
  if (i < n) o[i] = a[i] + b[i];
}

__global__ void k_zero_state(float* c, unsigned short* hbf)
{
  int i = blockIdx.x * blockDim.x + threadIdx.x;        // 4096
  if (i < TT * SED) { c[i] = 0.f; hbf[i] = 0; }
}

// fragment-ready swizzle of Whh1^T (bf16)
__global__ void k_wswz(const float* __restrict__ Whh1, unsigned short* __restrict__ W)
{
  int idx  = blockIdx.x * blockDim.x + threadIdx.x;     // < 1048576
  int e    = idx & 15;
  int lane = (idx >> 4) & 31;
  int kt   = (idx >> 9) & 15;
  int cb   = idx >> 13;                                 // 0..127
  int n = lane & 15, half = lane >> 4;
  int k = kt * 32 + half * 16 + e;
  int j = cb * 16 + n;
  W[idx] = f2bf(Whh1[(size_t)j * SED + k]);
}

// ---------------------------------------------------------------------------
// LSTM-1 step: 32 blocks x 1 wave; 4 gate tiles x 16 K-iters of WMMA + fused
// pointwise update.  (critical serial path: 1024 launches)
// ---------------------------------------------------------------------------
__global__ __launch_bounds__(32) void k_lstm1_step(
    const float* __restrict__ G1s,            // [8,2048] this step (bias folded)
    const unsigned short* __restrict__ Wswz,  // swizzled Whh1^T bf16
    const unsigned short* __restrict__ h_in,  // [8,512] bf16
    float* __restrict__ c,                    // [8,512] f32 (per-j owner)
    float* __restrict__ h_f,                  // [8,512] f32
    unsigned short* __restrict__ h_out)       // [8,512] bf16
{
  int b = blockIdx.x;                         // 0..31
  int lane = threadIdx.x;
  int n = lane & 15, half = lane >> 4;

  v8f acc[4];
  #pragma unroll
  for (int g = 0; g < 4; ++g) {
    int col = g * 512 + b * 16 + n;
    #pragma unroll
    for (int r = 0; r < 8; ++r)
      acc[g][r] = (half == 0) ? G1s[r * 2048 + col] : 0.f;
  }

  for (int kt = 0; kt < 16; ++kt) {
    F16 fa;                                   // A frag from h (rows 8..15 zero)
    if ((lane & 15) < 8) {
      const unsigned short* p = h_in + (lane & 15) * SED + kt * 32 + half * 16;
      fa.h[0] = *(const v8us*)p;
      fa.h[1] = *(const v8us*)(p + 8);
    } else {
      #pragma unroll
      for (int i = 0; i < 16; ++i) fa.u[i] = 0;
    }
    #pragma unroll
    for (int g = 0; g < 4; ++g) {
      int cb = g * 32 + b;
      F16 fb;
      const unsigned short* q = Wswz + ((size_t)(cb * 16 + kt) * 32 + lane) * 16;
      fb.h[0] = *(const v8us*)q;
      fb.h[1] = *(const v8us*)(q + 8);
      acc[g] = wmma_bf16(fa, fb, acc[g]);
    }
  }

  if (half == 0) {                            // lanes 0-15: update column j
    int j = b * 16 + n;
    #pragma unroll
    for (int r = 0; r < 8; ++r) {             // r == batch t
      float ig = sigm(acc[0][r]);
      float fg = sigm(acc[1][r]);
      float gg = tanhf(acc[2][r]);
      float og = sigm(acc[3][r]);
      float cn = fg * c[r * SED + j] + ig * gg;
      c[r * SED + j] = cn;
      float hn = og * tanhf(cn);
      h_f[r * SED + j] = hn;
      h_out[r * SED + j] = f2bf(hn);
    }
  }
}

__global__ void k_copy_h(const float* __restrict__ hf, float* __restrict__ fused)
{
  int i = blockIdx.x * blockDim.x + threadIdx.x;        // 4096
  if (i < TT * SED) fused[(i >> 9) * 1024 + (i & 511)] = hf[i];
}

// window = relu(fused @ Wf + bf)  : [8,1024]x[1024,512]
__global__ void k_window(const float* __restrict__ fused, const float* __restrict__ Wf,
                         const float* __restrict__ bfv, float* __restrict__ win)
{
  int u = blockIdx.x * blockDim.x + threadIdx.x;        // 4096
  int t = u >> 9, j = u & 511;
  float a = bfv[j];
  const float* x = fused + t * 1024;
  for (int v = 0; v < 1024; ++v) a += x[v] * Wf[v * 512 + j];
  win[t * 512 + j] = fmaxf(a, 0.f);
}

// ---------------------------------------------------------------------------
// LSTM-2 (8 steps, batch 1) + classifier + log_softmax, one workgroup.
// ---------------------------------------------------------------------------
__global__ __launch_bounds__(1024) void k_lstm2_head(
    const float* __restrict__ win, const float* __restrict__ Wih2,
    const float* __restrict__ Whh2, const float* __restrict__ bih2,
    const float* __restrict__ bhh2, const float* __restrict__ Wc,
    const float* __restrict__ bc, float* __restrict__ out)
{
  __shared__ float h[SED], cc[SED], gates[4 * SED], xs[SED], logits[16];
  int tid = threadIdx.x;
  if (tid < SED) { h[tid] = 0.f; cc[tid] = 0.f; }
  __syncthreads();

  for (int s = 0; s < 8; ++s) {
    if (tid < SED) xs[tid] = win[s * SED + tid];
    __syncthreads();
    for (int j = tid; j < 4 * SED; j += 1024) {
      float a = bih2[j] + bhh2[j];
      const float* wi = Wih2 + (size_t)j * SED;
      const float* wh = Whh2 + (size_t)j * SED;
      for (int k = 0; k < SED; ++k) a += xs[k] * wi[k] + h[k] * wh[k];
      gates[j] = a;
    }
    __syncthreads();
    if (tid < SED) {
      float ig = sigm(gates[tid]);
      float fg = sigm(gates[SED + tid]);
      float gg = tanhf(gates[2 * SED + tid]);
      float og = sigm(gates[3 * SED + tid]);
      float cn = fg * cc[tid] + ig * gg;
      cc[tid] = cn;
      h[tid] = og * tanhf(cn);
    }
    __syncthreads();
  }

  if (tid < 16) {
    float a = bc[tid];
    for (int k = 0; k < SED; ++k) a += h[k] * Wc[k * 16 + tid];
    logits[tid] = a;
  }
  __syncthreads();
  if (tid == 0) {
    float mx = -1e30f;
    for (int i = 0; i < 16; ++i) mx = fmaxf(mx, logits[i]);
    float se = 0.f;
    for (int i = 0; i < 16; ++i) se += expf(logits[i] - mx);
    float lse = mx + logf(se);
    for (int i = 0; i < 16; ++i) out[i] = logits[i] - lse;
  }
}

// ---------------------------------------------------------------------------
static void gemm(hipStream_t s, const unsigned short* A, const unsigned short* BT,
                 const float* bias, void* C, int M, int N, int K,
                 int lda, int ldbt, int ldc, int outMode, bool relu)
{
  dim3 g(N / 128, M / 128);
  k_gemm_async<<<g, 256, 0, s>>>(A, BT, bias, C, K, lda, ldbt, ldc,
                                 outMode, (int)relu);
}

extern "C" void kernel_launch(void* const* d_in, const int* in_sizes, int n_in,
                              void* d_out, int out_size, void* d_ws, size_t ws_size,
                              hipStream_t stream)
{
  (void)in_sizes; (void)n_in; (void)out_size; (void)ws_size;
  const float* adj  = (const float*)d_in[0];
  const float* att  = (const float*)d_in[1];
  const float* Wih1 = (const float*)d_in[2];
  const float* Whh1 = (const float*)d_in[3];
  const float* bih1 = (const float*)d_in[4];
  const float* bhh1 = (const float*)d_in[5];
  const float* W1   = (const float*)d_in[6];
  const float* b1   = (const float*)d_in[7];
  const float* W2   = (const float*)d_in[8];
  const float* b2   = (const float*)d_in[9];
  const float* Wf   = (const float*)d_in[10];
  const float* bfv  = (const float*)d_in[11];
  const float* Wih2 = (const float*)d_in[12];
  const float* Whh2 = (const float*)d_in[13];
  const float* bih2 = (const float*)d_in[14];
  const float* bhh2 = (const float*)d_in[15];
  const float* Wc   = (const float*)d_in[16];
  const float* bc   = (const float*)d_in[17];
  float* out = (float*)d_out;

  // workspace layout (float units), ~165 MB total
  float* w = (float*)d_ws;
  size_t off = 0;
  auto alloc = [&](size_t nf) { float* p = w + off; off += nf; return p; };
  float*          G1    = alloc(8192ull * 2048);                 // f32 gates
  unsigned short* X1bf  = (unsigned short*)alloc(8192ull * 1024);// [8192,2048] bf16
  unsigned short* ATTbf = (unsigned short*)alloc(8ull * 2048 * 512); // [8*2048,1024]
  unsigned short* WihBT = (unsigned short*)alloc(2048ull * 1024);// [2048,2048]
  unsigned short* W1T   = (unsigned short*)alloc(1024ull * 512); // [1024,1024]
  unsigned short* W2T   = (unsigned short*)alloc(512ull * 512);  // [512,1024]
  unsigned short* NTbf  = (unsigned short*)alloc(2048ull * 1024);// [2048,2048]
  unsigned short* TMP1T = (unsigned short*)alloc(1024ull * 1024);// [1024,2048]
  unsigned short* H1bf  = (unsigned short*)alloc(2048ull * 512); // [2048,1024]
  unsigned short* TMP2T = (unsigned short*)alloc(512ull * 1024); // [512,2048]
  unsigned short* H2T   = (unsigned short*)alloc(512ull * 1024); // [512,2048]
  float*          DINV  = alloc(2048);
  float*          BSUM  = alloc(2048);
  float*          FUSED = alloc(8 * 1024);
  float*          WIN   = alloc(8 * 512);
  float*          HF1   = alloc(8 * 512);
  float*          C1    = alloc(8 * 512);
  unsigned short* WSWZ  = (unsigned short*)alloc(524288);        // 1M bf16
  unsigned short* HBFA  = (unsigned short*)alloc(2048);
  unsigned short* HBFB  = (unsigned short*)alloc(2048);

  dim3 tb(32, 32);

  // ---- one-time bf16 operand prep ----
  k_biassum<<<8, 256, 0, stream>>>(bih1, bhh1, BSUM, 2048);
  k_wswz<<<4096, 256, 0, stream>>>(Whh1, WSWZ);
  k_zero_state<<<16, 256, 0, stream>>>(C1, HBFA);
  k_cvt<<<(2048 * 2048) / 256, 256, 0, stream>>>(Wih1, WihBT, 2048 * 2048);
  k_cvt<<<(8 * 2048 * 1024) / 256, 256, 0, stream>>>(att, ATTbf, 8 * 2048 * 1024);
  k_cvtT<<<dim3(32, 32), tb, 0, stream>>>(W1, W1T, 1024, 1024);
  k_cvtT<<<dim3(16, 32), tb, 0, stream>>>(W2, W2T, 1024, 512);
  k_transpose_att<<<dim3(64, 32, 8), tb, 0, stream>>>(att, X1bf);

  // ---- LSTM-1 precompute: G1 = X1 @ Wih1^T + (bih1+bhh1) ----
  gemm(stream, X1bf, WihBT, BSUM, G1, 8192, 2048, 2048, 2048, 2048, 2048, 0, false);

  // ---- GCN per temporal split ----
  for (int t = 0; t < TT; ++t) {
    const float* adjT = adj + (size_t)t * NN * NN;
    const unsigned short* xT = ATTbf + (size_t)t * NN * DD;
    k_deg<<<8, 256, 0, stream>>>(adjT, DINV);
    k_normT<<<dim3(64, 64), tb, 0, stream>>>(adjT, DINV, NTbf);
    // TMP1^T = (X W1)^T
    gemm(stream, xT, W1T, nullptr, TMP1T, 2048, 1024, 1024, 1024, 1024, 2048, 2, false);
    // H1 = relu(NT @ XW1 + b1)       (row-major bf16: next A operand)
    gemm(stream, NTbf, TMP1T, b1, H1bf, 2048, 1024, 2048, 2048, 2048, 1024, 1, true);
    // TMP2^T = (H1 W2)^T
    gemm(stream, H1bf, W2T, nullptr, TMP2T, 2048, 512, 1024, 1024, 1024, 2048, 2, false);
    // H2^T = relu(NT @ H1W2 + b2)^T  (bf16 [512,2048]: pool-friendly)
    gemm(stream, NTbf, TMP2T, b2, H2T, 2048, 512, 2048, 2048, 2048, 2048, 2, true);
    k_pool<<<2, 256, 0, stream>>>(H2T, FUSED + t * 1024 + 512);
  }

  // ---- LSTM-1 recurrence: 1024 dependent WMMA steps ----
  for (int d = 0; d < DD; ++d) {
    const unsigned short* hin = (d & 1) ? HBFB : HBFA;
    unsigned short* hout      = (d & 1) ? HBFA : HBFB;
    k_lstm1_step<<<32, 32, 0, stream>>>(G1 + (size_t)d * 8 * 2048, WSWZ,
                                        hin, C1, HF1, hout);
  }
  k_copy_h<<<16, 256, 0, stream>>>(HF1, FUSED);

  // ---- head ----
  k_window<<<16, 256, 0, stream>>>(FUSED, Wf, bfv, WIN);
  k_lstm2_head<<<1, 1024, 0, stream>>>(WIN, Wih2, Whh2, bih2, bhh2, Wc, bc, out);
}
